// RelativeGlobalAttention_65008624992654
// MI455X (gfx1250) — compile-verified, matrix operand order checked by reading, not run on hardware
//
#include <hip/hip_runtime.h>
#include <hip/hip_bf16.h>

typedef __attribute__((ext_vector_type(16))) __bf16 v16bf;
typedef __attribute__((ext_vector_type(8)))  float  v8f;

#define BATCH 2
#define NHEAD 8
#define SEQ   2048
#define EMBED 512
#define HD    64

// ---------------------------------------------------------------------------
// Fragment loaders for V_WMMA_F32_16X16X32_BF16 (wave32).
// A (16x32, 16-bit): lane<16 -> row=lane,   K = {k0..k0+7, k0+16..k0+23}
//                    lane>=16 -> row=lane-16,K = {k0+8..k0+15, k0+24..k0+31}
// B (32x16): symmetric layout with "row" = output column (load B^T row-major).
// ---------------------------------------------------------------------------
__device__ inline v16bf load_frag_f32(const float* __restrict__ base, int r0, int k0,
                                      int ld, int lane) {
  const int l = lane & 15, hi = (lane >> 4) & 1;
  const float* p = base + (size_t)(r0 + l) * ld + (k0 + hi * 8);
  v16bf f;
#pragma unroll
  for (int u = 0; u < 8; ++u) f[u] = (__bf16)p[u];
#pragma unroll
  for (int u = 0; u < 8; ++u) f[8 + u] = (__bf16)p[16 + u];
  return f;
}

__device__ inline v16bf load_frag_bf16(const __bf16* __restrict__ base, int r0, int k0,
                                       int ld, int lane) {
  const int l = lane & 15, hi = (lane >> 4) & 1;
  const __bf16* p = base + (size_t)(r0 + l) * ld + (k0 + hi * 8);
  v16bf f;
#pragma unroll
  for (int u = 0; u < 8; ++u) f[u] = p[u];
#pragma unroll
  for (int u = 0; u < 8; ++u) f[8 + u] = p[16 + u];
  return f;
}

__device__ inline v8f wmma_bf16(v16bf a, v16bf b, v8f c) {
  return __builtin_amdgcn_wmma_f32_16x16x32_bf16(false, a, false, b, (short)0, c,
                                                 false, false);
}

// ---------------------------------------------------------------------------
// Kernel 0: weight transpose->bf16 and reversed rel embedding.
// ---------------------------------------------------------------------------
__global__ void prep_kernel(const float* __restrict__ Wq, const float* __restrict__ Wk,
                            const float* __restrict__ Wv, const float* __restrict__ Wo,
                            const float* __restrict__ rel,
                            __bf16* __restrict__ Wqt, __bf16* __restrict__ Wkt,
                            __bf16* __restrict__ Wvt, __bf16* __restrict__ Wot,
                            __bf16* __restrict__ Rrev) {
  const int idx = blockIdx.x * blockDim.x + threadIdx.x;
  const int stride = gridDim.x * blockDim.x;
  for (int i = idx; i < EMBED * EMBED; i += stride) {
    int k = i / EMBED, n = i % EMBED;
    int t = n * EMBED + k;
    Wqt[t] = (__bf16)Wq[i];
    Wkt[t] = (__bf16)Wk[i];
    Wvt[t] = (__bf16)Wv[i];
    Wot[t] = (__bf16)Wo[i];
  }
  for (int i = idx; i < SEQ * HD; i += stride) {
    int t = i / HD, d = i % HD;
    Rrev[t * HD + d] = (__bf16)rel[(size_t)(SEQ - 1 - t) * HD + d];
  }
}

// ---------------------------------------------------------------------------
// Kernel 1/3: GEMM Y = X(4096x512,f32) @ Wt^T(bf16,[n][k]) + bias.
// Register blocking: each wave computes a 16x64 strip (4 N-tiles, 4 indep
// accumulators); A fragment reused 4x. Software-pipelined: iteration kc+1's
// A/B loads issue before iteration kc's WMMAs consume their fragments.
// layout 0: split-heads [b,h,s,d] bf16 ; layout 2: V transposed [b,h,d,s] bf16
// layout 3: plain row-major f32 (final output projection)
// ---------------------------------------------------------------------------
__global__ void proj_gemm_kernel(const float* __restrict__ X, const __bf16* __restrict__ Wt,
                                 const float* __restrict__ bias,
                                 __bf16* __restrict__ dstB, float* __restrict__ dstF,
                                 int layout) {
  const int lane = threadIdx.x & 31;
  const int wave = threadIdx.x >> 5;
  const int l = lane & 15, hi = (lane >> 4) & 1;
  const int task = blockIdx.x * 8 + wave;
  const int it = task >> 3;   // 0..255 M-tile
  const int nq = task & 7;    // 0..7  N-quad (covers 4 N-tiles = 64 cols)
  const int KC = EMBED / 32;  // 16 k-chunks

  v8f acc[4] = {};
  v16bf a_cur = load_frag_f32(X, it * 16, 0, EMBED, lane);
  v16bf b_cur[4];
#pragma unroll
  for (int u = 0; u < 4; ++u)
    b_cur[u] = load_frag_bf16(Wt, (nq * 4 + u) * 16, 0, EMBED, lane);

#pragma unroll 4
  for (int kc = 0; kc < KC; ++kc) {
    v16bf a_nxt;
    v16bf b_nxt[4];
    if (kc + 1 < KC) {
      __builtin_prefetch(X + (size_t)(it * 16 + l) * EMBED + (kc + 2) * 32, 0, 3);
      a_nxt = load_frag_f32(X, it * 16, (kc + 1) * 32, EMBED, lane);
#pragma unroll
      for (int u = 0; u < 4; ++u)
        b_nxt[u] = load_frag_bf16(Wt, (nq * 4 + u) * 16, (kc + 1) * 32, EMBED, lane);
    }
#pragma unroll
    for (int u = 0; u < 4; ++u)
      acc[u] = wmma_bf16(a_cur, b_cur[u], acc[u]);
    if (kc + 1 < KC) {
      a_cur = a_nxt;
#pragma unroll
      for (int u = 0; u < 4; ++u) b_cur[u] = b_nxt[u];
    }
  }

#pragma unroll
  for (int u = 0; u < 4; ++u) {
    const int n = (nq * 4 + u) * 16 + l;
    const float bv = bias[n];
    const int h = n >> 6, d = n & 63;
#pragma unroll
    for (int r = 0; r < 8; ++r) {
      const int gi = it * 16 + r + hi * 8;     // global row in [0,4096)
      const float v = acc[u][r] + bv;
      const int bidx = gi >> 11, s = gi & (SEQ - 1);
      if (layout == 0) {
        dstB[(((size_t)(bidx * NHEAD + h)) * SEQ + s) * HD + d] = (__bf16)v;
      } else if (layout == 2) {
        dstB[(((size_t)(bidx * NHEAD + h)) * HD + d) * SEQ + s] = (__bf16)v;
      } else {
        dstF[(size_t)gi * EMBED + n] = v;
      }
    }
  }
}

// ---------------------------------------------------------------------------
// Kernel 2: fused attention for one (b,h, 32-row q-block).
// LDS holds logits strip [32][2048] f32 (256 KB, dynamic).
// Phase A: scores = Q K^T (WMMA, pipelined K stream) -> LDS
// Phase B: P = Q Rrev^T (WMMA), scatter-add P[i,t] -> LDS[i, i-t]  (skew)
// Phase C: softmax rows (scale 1/8), float4-vectorized, write attn to d_out
// Phase D: O = attn @ V (WMMA, A from LDS f32->bf16, pipelined V stream)
// ---------------------------------------------------------------------------
__global__ void attn_kernel(const __bf16* __restrict__ Qh, const __bf16* __restrict__ Kh,
                            const __bf16* __restrict__ Vt, const __bf16* __restrict__ Rrev,
                            float* __restrict__ attnOut, float* __restrict__ Obuf) {
  extern __shared__ float sL[];                 // [32][SEQ]
  const int lane = threadIdx.x & 31;
  const int wave = threadIdx.x >> 5;            // 8 waves
  const int l = lane & 15, hi = (lane >> 4) & 1;
  const int bh = blockIdx.x >> 6;               // 0..15
  const int qb = blockIdx.x & 63;               // 0..63
  const int i0 = qb * 32;
  const __bf16* Q = Qh + (size_t)bh * SEQ * HD;
  const __bf16* K = Kh + (size_t)bh * SEQ * HD;
  const __bf16* V = Vt + (size_t)bh * HD * SEQ;

  // Q fragments for both 16-row tiles, both 32-wide K chunks of HD=64.
  v16bf aq[2][2];
#pragma unroll
  for (int it = 0; it < 2; ++it)
#pragma unroll
    for (int kc = 0; kc < 2; ++kc)
      aq[it][kc] = load_frag_bf16(Q, i0 + it * 16, kc * 32, HD, lane);

  // ---- Phase A: scores -> LDS (pipelined K fragment stream) ----
  {
    v16bf b0 = load_frag_bf16(K, wave * 16, 0, HD, lane);
    v16bf b1 = load_frag_bf16(K, wave * 16, 32, HD, lane);
    for (int jt = wave; jt < SEQ / 16; jt += 8) {
      v16bf n0, n1;
      if (jt + 8 < SEQ / 16) {
        __builtin_prefetch(K + ((size_t)(jt + 16) * 16 + l) * HD, 0, 3);
        n0 = load_frag_bf16(K, (jt + 8) * 16, 0, HD, lane);
        n1 = load_frag_bf16(K, (jt + 8) * 16, 32, HD, lane);
      }
#pragma unroll
      for (int it = 0; it < 2; ++it) {
        v8f c = {};
        c = wmma_bf16(aq[it][0], b0, c);
        c = wmma_bf16(aq[it][1], b1, c);
#pragma unroll
        for (int r = 0; r < 8; ++r)
          sL[(it * 16 + r + hi * 8) * SEQ + jt * 16 + l] = c[r];
      }
      if (jt + 8 < SEQ / 16) { b0 = n0; b1 = n1; }
    }
  }
  __syncthreads();

  // ---- Phase B: relative scores, skew scatter-add ----
  const int numT = i0 / 16 + 2;                 // t-tiles covering t <= i0+31
  for (int tt = wave; tt < numT; tt += 8) {
    v16bf b0 = load_frag_bf16(Rrev, tt * 16, 0, HD, lane);
    v16bf b1 = load_frag_bf16(Rrev, tt * 16, 32, HD, lane);
#pragma unroll
    for (int it = 0; it < 2; ++it) {
      v8f c = {};
      c = wmma_bf16(aq[it][0], b0, c);
      c = wmma_bf16(aq[it][1], b1, c);
#pragma unroll
      for (int r = 0; r < 8; ++r) {
        const int il = it * 16 + r + hi * 8;
        const int gi = i0 + il;
        const int t = tt * 16 + l;
        if (t <= gi) sL[il * SEQ + (gi - t)] += c[r];  // unique (il,j): no race
      }
    }
  }
  __syncthreads();

  // ---- Phase C: softmax + write attn (float4 vectorized) ----
  const float scale = 0.125f;                   // 1/sqrt(64)
#pragma unroll
  for (int rr = 0; rr < 4; ++rr) {
    const int il = wave * 4 + rr;
    const int gi = i0 + il;
    float4* row4 = (float4*)(sL + il * SEQ);    // 16B-aligned (row = 8KB)
    float m = -1e30f;
    for (int c = lane; c < SEQ / 4; c += 32) {
      const float4 v = row4[c];
      m = fmaxf(m, fmaxf(fmaxf(v.x, v.y), fmaxf(v.z, v.w)));
    }
    m *= scale;
#pragma unroll
    for (int off = 16; off > 0; off >>= 1) m = fmaxf(m, __shfl_xor(m, off, 32));
    float s = 0.f;
    for (int c = lane; c < SEQ / 4; c += 32) {
      float4 v = row4[c];
      v.x = __expf(v.x * scale - m);
      v.y = __expf(v.y * scale - m);
      v.z = __expf(v.z * scale - m);
      v.w = __expf(v.w * scale - m);
      row4[c] = v;
      s += (v.x + v.y) + (v.z + v.w);
    }
#pragma unroll
    for (int off = 16; off > 0; off >>= 1) s += __shfl_xor(s, off, 32);
    const float inv = 1.f / s;
    float4* arow4 = (float4*)(attnOut + ((size_t)bh * SEQ + gi) * SEQ);
    for (int c = lane; c < SEQ / 4; c += 32) {
      float4 v = row4[c];
      v.x *= inv; v.y *= inv; v.z *= inv; v.w *= inv;
      row4[c] = v;
      arow4[c] = v;                             // attn written exactly once (b128)
    }
  }
  __syncthreads();

  // ---- Phase D: O = attn @ V (pipelined V fragment stream) ----
  const int it = wave >> 2;                     // 0..1
  const int ntl = wave & 3;                     // 0..3 (HD/16)
  v8f acc = {};
  {
    v16bf b_cur = load_frag_bf16(V, ntl * 16, 0, SEQ, lane);
    for (int kt = 0; kt < SEQ / 32; ++kt) {
      v16bf b_nxt;
      if (kt + 1 < SEQ / 32) {
        __builtin_prefetch(V + (size_t)(ntl * 16 + l) * SEQ + (kt + 2) * 32, 0, 3);
        b_nxt = load_frag_bf16(V, ntl * 16, (kt + 1) * 32, SEQ, lane);
      }
      v16bf a = load_frag_f32(sL, it * 16, kt * 32, SEQ, lane);
      acc = wmma_bf16(a, b_cur, acc);
      if (kt + 1 < SEQ / 32) b_cur = b_nxt;
    }
  }
  const int bb = bh >> 3, h = bh & 7;
  const int d = ntl * 16 + l;
#pragma unroll
  for (int r = 0; r < 8; ++r) {
    const int il = it * 16 + r + hi * 8;
    const int gi = i0 + il;
    Obuf[((size_t)(bb * SEQ + gi)) * EMBED + h * HD + d] = acc[r];
  }
}

// ---------------------------------------------------------------------------
extern "C" void kernel_launch(void* const* d_in, const int* in_sizes, int n_in,
                              void* d_out, int out_size, void* d_ws, size_t ws_size,
                              hipStream_t stream) {
  const float* q_in = (const float*)d_in[0];
  const float* k_in = (const float*)d_in[1];
  const float* v_in = (const float*)d_in[2];
  const float* Wq   = (const float*)d_in[3];
  const float* bq   = (const float*)d_in[4];
  const float* Wk   = (const float*)d_in[5];
  const float* bk   = (const float*)d_in[6];
  const float* Wv   = (const float*)d_in[7];
  const float* bv   = (const float*)d_in[8];
  const float* Wo   = (const float*)d_in[9];
  const float* bo   = (const float*)d_in[10];
  const float* rel  = (const float*)d_in[11];

  char* ws = (char*)d_ws;
  size_t off = 0;
  __bf16* Wqt  = (__bf16*)(ws + off); off += (size_t)EMBED * EMBED * 2;
  __bf16* Wkt  = (__bf16*)(ws + off); off += (size_t)EMBED * EMBED * 2;
  __bf16* Wvt  = (__bf16*)(ws + off); off += (size_t)EMBED * EMBED * 2;
  __bf16* Wot  = (__bf16*)(ws + off); off += (size_t)EMBED * EMBED * 2;
  __bf16* Rrev = (__bf16*)(ws + off); off += (size_t)SEQ * HD * 2;
  __bf16* Qh   = (__bf16*)(ws + off); off += (size_t)BATCH * NHEAD * SEQ * HD * 2;
  __bf16* Kh   = (__bf16*)(ws + off); off += (size_t)BATCH * NHEAD * SEQ * HD * 2;
  __bf16* Vtb  = (__bf16*)(ws + off); off += (size_t)BATCH * NHEAD * SEQ * HD * 2;
  float*  Obuf = (float*)(ws + off);  off += (size_t)BATCH * SEQ * EMBED * 4;

  float* out1    = (float*)d_out;
  float* attnOut = out1 + (size_t)BATCH * SEQ * EMBED;

  prep_kernel<<<512, 256, 0, stream>>>(Wq, Wk, Wv, Wo, rel, Wqt, Wkt, Wvt, Wot, Rrev);

  proj_gemm_kernel<<<256, 256, 0, stream>>>(q_in, Wqt, bq, Qh,  nullptr, 0);
  proj_gemm_kernel<<<256, 256, 0, stream>>>(k_in, Wkt, bk, Kh,  nullptr, 0);
  proj_gemm_kernel<<<256, 256, 0, stream>>>(v_in, Wvt, bv, Vtb, nullptr, 2);

  const size_t ldsBytes = (size_t)32 * SEQ * sizeof(float);   // 256 KB
  (void)hipFuncSetAttribute((const void*)attn_kernel,
                            hipFuncAttributeMaxDynamicSharedMemorySize,
                            (int)ldsBytes);
  attn_kernel<<<BATCH * NHEAD * (SEQ / 32), 256, ldsBytes, stream>>>(
      Qh, Kh, Vtb, Rrev, attnOut, Obuf);

  proj_gemm_kernel<<<256, 256, 0, stream>>>(Obuf, Wot, bo, nullptr, out1, 3);
}
